// CrossAttention_39444979646641
// MI455X (gfx1250) — matrix operand order repaired and checked by reading, not compiled, not run on hardware
//
#include <hip/hip_runtime.h>

typedef __attribute__((ext_vector_type(16))) __bf16 v16bf;
typedef __attribute__((ext_vector_type(8)))  float  v8f;

struct alignas(16) U128 { unsigned int x0, x1, x2, x3; };
union  Frag16 { v16bf v; U128 u[2]; };

#define NH   8
#define HD   64
#define CDIM 512
#define NTOK 1024
#define BSZ  4

// ---------------- f32 -> bf16 convert ----------------
__global__ __launch_bounds__(256) void cvt_bf16(const float* __restrict__ src,
                                                __bf16* __restrict__ dst, int n) {
  int i = blockIdx.x * 256 + threadIdx.x;
  if (i < n) dst[i] = (__bf16)src[i];
}

// ---------------- W[k][n] -> WT[n][k] bf16 ----------------
__global__ __launch_bounds__(256) void transpose_w_bf16(const float* __restrict__ W,
                                                        __bf16* __restrict__ WT) {
  int i = blockIdx.x * 256 + threadIdx.x;   // 0 .. 512*512-1
  int n = i >> 9, k = i & 511;
  WT[n * CDIM + k] = (__bf16)W[k * CDIM + n];
}

// ---------------- projection GEMM: Out = X @ W + b (bf16 in, bf16 out, f32 acc) ----
// each wave: one 16(M) x 32(N) strip = 2 WMMA tiles sharing the A fragment
__global__ __launch_bounds__(256) void proj_gemm(const __bf16* __restrict__ X,
                                                 const __bf16* __restrict__ WT,
                                                 const float* __restrict__ bias,
                                                 __bf16* __restrict__ Out) {
  const int lane = threadIdx.x & 31;
  const int wave = threadIdx.x >> 5;
  const int job  = blockIdx.x * 8 + wave;   // 4096 jobs
  const int tm   = job >> 4;                // 0..255  (M tiles: 4096/16)
  const int tnp  = job & 15;                // 0..15   (N strips: 512/32)
  const int col  = lane & 15;
  const int hi   = (lane >> 4) & 1;

  const __bf16* arow  = X  + (size_t)(tm * 16 + col) * CDIM;        // A row
  const __bf16* brow0 = WT + (size_t)(tnp * 32 + col) * CDIM;       // B cols 0-15
  const __bf16* brow1 = WT + (size_t)(tnp * 32 + 16 + col) * CDIM;  // B cols 16-31

  v8f acc0 = {}, acc1 = {};
#pragma unroll
  for (int kk = 0; kk < CDIM / 32; ++kk) {
    Frag16 a, b0, b1;
    const int ka = kk * 32 + hi * 16;       // A: lanes>=16 hold K 16-31
    a.u[0] = *(const U128*)(arow + ka);
    a.u[1] = *(const U128*)(arow + ka + 8);
    const int kb = kk * 32 + hi * 16;       // B: lanes>=16 hold K 16-31
    b0.u[0] = *(const U128*)(brow0 + kb);
    b0.u[1] = *(const U128*)(brow0 + kb + 8);
    b1.u[0] = *(const U128*)(brow1 + kb);
    b1.u[1] = *(const U128*)(brow1 + kb + 8);
    acc0 = __builtin_amdgcn_wmma_f32_16x16x32_bf16(false, a.v, false, b0.v,
                                                   (short)0, acc0, false, false);
    acc1 = __builtin_amdgcn_wmma_f32_16x16x32_bf16(false, a.v, false, b1.v,
                                                   (short)0, acc1, false, false);
  }
  const int nj0 = tnp * 32 + col;
  const int nj1 = nj0 + 16;
  const float bv0 = bias[nj0];
  const float bv1 = bias[nj1];
  const int rowbase = tm * 16 + hi * 8;
#pragma unroll
  for (int r = 0; r < 8; ++r) {
    Out[(size_t)(rowbase + r) * CDIM + nj0] = (__bf16)(acc0[r] + bv0);
    Out[(size_t)(rowbase + r) * CDIM + nj1] = (__bf16)(acc1[r] + bv1);
  }
}

// ---------------- windowed cross-attention ----------------
// block = one (b, head, patch); 8 waves x 8 query-chunks of 16 queries
__global__ __launch_bounds__(256) void xattn(const __bf16* __restrict__ Qb,
                                             const __bf16* __restrict__ Kb,
                                             const __bf16* __restrict__ Vb,
                                             float* __restrict__ Out) {
  __shared__ alignas(16) __bf16 sK[16 * 64];
  __shared__ alignas(16) __bf16 sV[16 * 64];
  __shared__ alignas(16) __bf16 sP[8][16 * 16];

  const int bid = blockIdx.x;              // (b*8+h)*64 + p
  const int p  = bid & 63;
  const int h  = (bid >> 6) & 7;
  const int b  = bid >> 9;
  const int py = p >> 3, px = p & 7;

  { // async global->LDS load of the 16x64 K and V patch tiles (8B per lane)
    const int t  = threadIdx.x;
    const int s  = t >> 4;                 // key slot 0..15
    const int d  = (t & 15) * 4;           // 4 bf16 per thread
    const int dy = s >> 2, dx = s & 3;
    const int n  = (py * 4 + dy) * 32 + (px * 4 + dx);
    const size_t src = ((size_t)(b * NTOK + n)) * CDIM + h * HD + d;
    const __bf16* gk = Kb + src;
    const __bf16* gv = Vb + src;
    const unsigned ldsK = (unsigned)(unsigned long long)(const void*)&sK[s * 64 + d];
    const unsigned ldsV = (unsigned)(unsigned long long)(const void*)&sV[s * 64 + d];
    asm volatile("global_load_async_to_lds_b64 %0, %2, off\n\t"
                 "global_load_async_to_lds_b64 %1, %3, off"
                 :: "v"(ldsK), "v"(ldsV), "v"(gk), "v"(gv)
                 : "memory");
    asm volatile("s_wait_asynccnt 0x0" ::: "memory");
  }
  __syncthreads();

  const int lane = threadIdx.x & 31;
  const int wave = threadIdx.x >> 5;
  const int col = lane & 15;
  const int hi  = (lane >> 4) & 1;

  // K^T as B-fragments for S = Q*K^T (K dim = 64 -> 2 steps); reused all chunks
  Frag16 fK[2];
#pragma unroll
  for (int st = 0; st < 2; ++st) {
    const int kb = st * 32 + hi * 16;
    fK[st].u[0] = *(const U128*)(&sK[col * 64 + kb]);
    fK[st].u[1] = *(const U128*)(&sK[col * 64 + kb + 8]);
  }
  // V as B-fragments for O = P*V (K padded 16 -> 32 with zeros); reused all chunks
  Frag16 fV[4];
#pragma unroll
  for (int nt = 0; nt < 4; ++nt) {
#pragma unroll
    for (int i = 0; i < 16; ++i)
      fV[nt].v[i] = (i >= 8) ? (__bf16)0.0f : sV[(hi * 8 + i) * 64 + nt * 16 + col];
  }

  __bf16* myP = sP[wave];
  const size_t obase = (size_t)bid * (NTOK * HD);

  for (int it = 0; it < 8; ++it) {
    const int chunk = wave * 8 + it;       // 0..63
    const int q = chunk * 16 + col;
    const __bf16* qrow = Qb + (size_t)q * CDIM + h * HD;

    v8f acc = {};
#pragma unroll
    for (int st = 0; st < 2; ++st) {
      Frag16 a;
      const int ka = st * 32 + hi * 16;
      a.u[0] = *(const U128*)(qrow + ka);
      a.u[1] = *(const U128*)(qrow + ka + 8);
      acc = __builtin_amdgcn_wmma_f32_16x16x32_bf16(false, a.v, false, fK[st].v,
                                                    (short)0, acc, false, false);
    }

    // softmax over the 16 key slots: each row lives in 16 lanes of a half-wave
    const int rowbase = hi * 8;
#pragma unroll
    for (int r = 0; r < 8; ++r) {
      float v = acc[r] * 0.125f;           // HEAD_DIM^-0.5
      float m = v;
      m = fmaxf(m, __shfl_xor(m, 1, 32));
      m = fmaxf(m, __shfl_xor(m, 2, 32));
      m = fmaxf(m, __shfl_xor(m, 4, 32));
      m = fmaxf(m, __shfl_xor(m, 8, 32));
      float e = __expf(v - m);
      float s = e;
      s += __shfl_xor(s, 1, 32);
      s += __shfl_xor(s, 2, 32);
      s += __shfl_xor(s, 4, 32);
      s += __shfl_xor(s, 8, 32);
      myP[(rowbase + r) * 16 + col] = (__bf16)(e / s);
    }

    // reload P as an A-fragment (16x16 real, K 16..31 zero-padded);
    // same-wave LDS ops are in-order, no barrier needed
    Frag16 ap;
    ap.u[0] = *(const U128*)(&myP[col * 16 + hi * 8]);
    ap.u[1].x0 = ap.u[1].x1 = ap.u[1].x2 = ap.u[1].x3 = 0u;

#pragma unroll
    for (int nt = 0; nt < 4; ++nt) {
      v8f o = {};
      o = __builtin_amdgcn_wmma_f32_16x16x32_bf16(false, ap.v, false, fV[nt].v,
                                                  (short)0, o, false, false);
#pragma unroll
      for (int r = 0; r < 8; ++r)
        Out[obase + (size_t)(chunk * 16 + rowbase + r) * HD + nt * 16 + col] = o[r];
    }
  }
}

extern "C" void kernel_launch(void* const* d_in, const int* in_sizes, int n_in,
                              void* d_out, int out_size, void* d_ws, size_t ws_size,
                              hipStream_t stream) {
  (void)in_sizes; (void)n_in; (void)out_size; (void)ws_size;
  const float* x_s = (const float*)d_in[0];
  const float* x_l = (const float*)d_in[1];
  const float* Wq  = (const float*)d_in[2];
  const float* bq  = (const float*)d_in[3];
  const float* Wk  = (const float*)d_in[4];
  const float* bk  = (const float*)d_in[5];
  const float* Wv  = (const float*)d_in[6];
  const float* bv  = (const float*)d_in[7];

  __bf16* wsb = (__bf16*)d_ws;
  const size_t NXE = (size_t)BSZ * NTOK * CDIM;   // 2,097,152 elems
  const size_t NWE = (size_t)CDIM * CDIM;         //   262,144 elems
  __bf16* Xl  = wsb;
  __bf16* Xs  = Xl  + NXE;
  __bf16* WqT = Xs  + NXE;
  __bf16* WkT = WqT + NWE;
  __bf16* WvT = WkT + NWE;
  __bf16* Qb  = WvT + NWE;
  __bf16* Kb  = Qb  + NXE;
  __bf16* Vb  = Kb  + NXE;                        // total ws use ~21.5 MiB

  cvt_bf16<<<(int)(NXE / 256), 256, 0, stream>>>(x_l, Xl, (int)NXE);
  cvt_bf16<<<(int)(NXE / 256), 256, 0, stream>>>(x_s, Xs, (int)NXE);
  transpose_w_bf16<<<(int)(NWE / 256), 256, 0, stream>>>(Wq, WqT);
  transpose_w_bf16<<<(int)(NWE / 256), 256, 0, stream>>>(Wk, WkT);
  transpose_w_bf16<<<(int)(NWE / 256), 256, 0, stream>>>(Wv, WvT);
  proj_gemm<<<512, 256, 0, stream>>>(Xl, WqT, bq, Qb);
  proj_gemm<<<512, 256, 0, stream>>>(Xs, WkT, bk, Kb);
  proj_gemm<<<512, 256, 0, stream>>>(Xs, WvT, bv, Vb);
  xattn<<<2048, 256, 0, stream>>>(Qb, Kb, Vb, (float*)d_out);
}
